// HPUMLAImpl_23699629539460
// MI455X (gfx1250) — compile-verified, matrix operand order checked by reading, not run on hardware
//
#include <hip/hip_runtime.h>
#include <hip/hip_bf16.h>

#define NB   32      // batch
#define NH   16      // heads
#define LC   512     // latent dim
#define DN   128     // nope dim
#define DR   64      // rope dim
#define DV   128     // value dim
#define DQK  192     // DN+DR
#define DK   576     // LC+DR
#define BS   128     // cache block size
#define BPS  32      // blocks per seq
#define NCH  (DK / 32)   // 18 K-chunks for scores
#define SCALE_F 0.07216878364870323f   // 192^-0.5
#define NEG_INF (-__builtin_huge_valf())

typedef __attribute__((ext_vector_type(16))) __bf16 v16bf;
typedef __attribute__((ext_vector_type(8)))  float  v8f;
typedef __attribute__((ext_vector_type(4)))  __bf16 bf16x4;
typedef __attribute__((ext_vector_type(8)))  __bf16 bf16x8;

// ISA 7.12.2: 16-bit A-matrix 16x32 slot->K mapping (B uses same, column-major)
__device__ __forceinline__ int kslot(int lane, int j) {
    int i = j >> 1, par = j & 1;
    int hi = (lane >= 16) ? 8 : 0;
    return (i < 4) ? (hi + 2 * i + par) : (16 + hi + 2 * (i - 4) + par);
}

// ---------------- Kernel 1: projections + RoPE -> bf16 query & latent ----------------
__global__ __launch_bounds__(256) void prep_kernel(
    const float* __restrict__ q, const float* __restrict__ k_c,
    const float* __restrict__ k_pe, const float* __restrict__ W_UK_T,
    const int* __restrict__ input_positions,
    __bf16* __restrict__ query_ws, __bf16* __restrict__ latent_ws) {
    int b = blockIdx.x, h = blockIdx.y, t = threadIdx.x;
    int pos = input_positions[b];
    const float* qn = q + (size_t)(b * NH + h) * DQK;
    const float* W  = W_UK_T + (size_t)h * DN * LC;
    for (int l = t; l < LC; l += 256) {
        float acc = 0.f;
        for (int d = 0; d < DN; ++d) acc += qn[d] * W[(size_t)d * LC + l];
        query_ws[(size_t)(b * NH + h) * DK + l] = (__bf16)acc;
    }
    if (t < DR) {
        const int half = DR / 2;
        int j = t % half;
        float inv = __powf(10000.f, -(float)j / (float)half);
        float f = (float)pos * inv;
        float c = __cosf(f), s = __sinf(f);
        float x   = qn[DN + t];
        float rot = (t < half) ? -qn[DN + t + half] : qn[DN + t - half];
        query_ws[(size_t)(b * NH + h) * DK + LC + t] = (__bf16)(x * c + rot * s);
    }
    if (h == 0) {
        for (int i = t; i < LC; i += 256)
            latent_ws[(size_t)b * DK + i] = (__bf16)k_c[(size_t)b * LC + i];
        if (t < DR) {
            const int half = DR / 2;
            int j = t % half;
            float inv = __powf(10000.f, -(float)j / (float)half);
            float f = (float)pos * inv;
            float c = __cosf(f), s = __sinf(f);
            float x   = k_pe[(size_t)b * DR + t];
            float rot = (t < half) ? -k_pe[(size_t)b * DR + t + half]
                                   : k_pe[(size_t)b * DR + t - half];
            latent_ws[(size_t)b * DK + LC + t] = (__bf16)(x * c + rot * s);
        }
    }
}

// -------- staging helpers: split issue(load) / commit(cvt+store) for pipelining.
// Current-token override is done by VALUE SELECTION (branchless in commit): the
// threads whose segments lie in the override row each pre-load their latent bf16x4.
struct KStage { float4 v[4]; bf16x4 latv; int curS; };
struct VStage { float4 v[8]; bf16x4 latv; int curS; };

__device__ __forceinline__ void k_issue(const float* __restrict__ kbase,
                                        const __bf16* __restrict__ latb,
                                        int c, int t, bool haveCur, int curRow,
                                        KStage& st) {
    #pragma unroll
    for (int s = 0; s < 4; ++s) {
        int seg = t + (s << 8);
        int p = seg >> 3, kk = (seg & 7) << 2;        // p = (t>>3) + 32s
        st.v[s] = *(const float4*)(kbase + (size_t)p * DK + c * 32 + kk);
    }
    st.curS = -1;
    if (haveCur && ((t >> 3) == (curRow & 31))) {
        st.curS = curRow >> 5;
        st.latv = *(const bf16x4*)(latb + c * 32 + ((t & 7) << 2));
    }
}
__device__ __forceinline__ void k_commit(__bf16* __restrict__ dst, int t,
                                         const KStage& st) {
    #pragma unroll
    for (int s = 0; s < 4; ++s) {
        int seg = t + (s << 8);
        int p = seg >> 3, kk = (seg & 7) << 2;
        bf16x4 o;
        o[0] = (__bf16)st.v[s].x; o[1] = (__bf16)st.v[s].y;
        o[2] = (__bf16)st.v[s].z; o[3] = (__bf16)st.v[s].w;
        o = (s == st.curS) ? st.latv : o;             // cndmask, no exec games
        *(bf16x4*)(dst + p * 32 + kk) = o;
    }
}
// V half-chunk: rows pc*32..+32, dims half*256..+256 (it = pc*2+half)
__device__ __forceinline__ void v_issue(const float* __restrict__ kbase,
                                        const __bf16* __restrict__ latb,
                                        int it, int t, bool haveCur, int curRow,
                                        VStage& st) {
    int pc = it >> 1, half = it & 1;
    #pragma unroll
    for (int s = 0; s < 8; ++s) {
        int seg = t + (s << 8);
        int r = seg >> 6, d = (seg & 63) << 2;        // r = (t>>6) + 4s
        st.v[s] = *(const float4*)(kbase + (size_t)(pc * 32 + r) * DK + half * 256 + d);
    }
    st.curS = -1;
    bool curHere = haveCur && ((curRow >> 5) == pc);
    int curR = curRow & 31;
    if (curHere && ((t >> 6) == (curR & 3))) {
        st.curS = curR >> 2;
        st.latv = *(const bf16x4*)(latb + half * 256 + ((t & 63) << 2));
    }
}
__device__ __forceinline__ void v_commit(__bf16* __restrict__ dst, int t,
                                         const VStage& st) {
    #pragma unroll
    for (int s = 0; s < 8; ++s) {
        int seg = t + (s << 8);
        int r = seg >> 6, d = (seg & 63) << 2;
        bf16x4 o;
        o[0] = (__bf16)st.v[s].x; o[1] = (__bf16)st.v[s].y;
        o[2] = (__bf16)st.v[s].z; o[3] = (__bf16)st.v[s].w;
        o = (s == st.curS) ? st.latv : o;
        *(bf16x4*)(dst + r * 256 + d) = o;
    }
}

// ---------------- Kernel 2: split-KV flash decode with bf16 WMMA ----------------
__global__ __launch_bounds__(256) void attn_partial_kernel(
    const float* __restrict__ kv_cache, const int* __restrict__ block_tables,
    const int* __restrict__ seq_lens,
    const __bf16* __restrict__ query_ws, const __bf16* __restrict__ latent_ws,
    float* __restrict__ ctx_part, float* __restrict__ m_part,
    float* __restrict__ l_part) {
    __shared__ alignas(16) __bf16 q_s[NH * DK];     // 18432 B
    __shared__ alignas(16) float  s_s[NH * BS];     //  8192 B
    __shared__ alignas(16) __bf16 p_s[NH * BS];     //  4096 B
    __shared__ alignas(16) __bf16 kv_s[16384];      // 32768 B (2x8KB K bufs / 2x16KB V bufs)
    __shared__ float red_s[NH * 8];                 //   512 B
    __shared__ float m_sh[NH];

    int b = blockIdx.x, blk = blockIdx.y;
    int t = threadIdx.x;
    int wave = t >> 5, lane = t & 31;
    int seq  = seq_lens[b];
    int base = blk * BS;
    float* ctxp = ctx_part + (size_t)((b * BPS + blk) * NH) * LC;

    if (base >= seq) { // fully masked split
        for (int i = t; i < NH * LC; i += 256) ctxp[i] = 0.f;
        if (t < NH) {
            m_part[(b * BPS + blk) * NH + t] = NEG_INF;
            l_part[(b * BPS + blk) * NH + t] = 0.f;
        }
        return;
    }
    int nvalid = (seq - base < BS) ? (seq - base) : BS;
    int cur = seq - 1;
    bool haveCur = (cur >> 7) == blk;   // current-token row lives in this block
    int curRow = cur & (BS - 1);
    int phys = block_tables[b * BPS + blk];
    const float* kbase = kv_cache + (size_t)phys * BS * DK;
    const __bf16* latb = latent_ws + (size_t)b * DK;

    // prime K pipeline (issue chunk-0 loads before q copy so latency overlaps)
    KStage kst;
    k_issue(kbase, latb, 0, t, haveCur, curRow, kst);
    {   // q -> LDS, 16B vector moves
        const bf16x8* qsrc = (const bf16x8*)(query_ws + (size_t)b * NH * DK);
        bf16x8* qdst = (bf16x8*)q_s;
        for (int i = t; i < NH * DK / 8; i += 256) qdst[i] = qsrc[i];
    }
    k_commit(kv_s, t, kst);
    __syncthreads();

    int row = lane & 15;
    int mo  = (lane >= 16) ? 8 : 0;

    // ---- scores: D(16 heads x 16 pos per wave), 18 chunks, double-buffered ----
    v8f acc = {};
    for (int c = 0; c < NCH; ++c) {
        const __bf16* cbuf = kv_s + (c & 1) * (BS * 32);
        KStage nx;
        if (c + 1 < NCH) k_issue(kbase, latb, c + 1, t, haveCur, curRow, nx);
        v16bf a, bm;
        #pragma unroll
        for (int j = 0; j < 16; ++j) {
            int kk = kslot(lane, j);
            a[j]  = q_s[row * DK + c * 32 + kk];
            bm[j] = cbuf[(wave * 16 + row) * 32 + kk];
        }
        acc = __builtin_amdgcn_wmma_f32_16x16x32_bf16(false, a, false, bm,
                                                      (short)0, acc, false, false);
        if (c + 1 < NCH) k_commit(kv_s + ((c + 1) & 1) * (BS * 32), t, nx);
        __syncthreads();
    }
    #pragma unroll
    for (int r = 0; r < 8; ++r)
        s_s[(r + mo) * BS + wave * 16 + row] = acc[r] * SCALE_F;
    __syncthreads();

    // ---- per-split softmax (parallel two-level reduction) ----
    if (t < 128) {
        int h = t >> 3, seg = t & 7;
        float m = NEG_INF;
        for (int p = seg * 16; p < seg * 16 + 16; ++p)
            if (p < nvalid) m = fmaxf(m, s_s[h * BS + p]);
        red_s[h * 8 + seg] = m;
    }
    __syncthreads();
    if (t < NH) {
        float m = NEG_INF;
        for (int i = 0; i < 8; ++i) m = fmaxf(m, red_s[t * 8 + i]);
        m_sh[t] = m;
        m_part[(b * BPS + blk) * NH + t] = m;
    }
    __syncthreads();
    if (t < 128) {
        int h = t >> 3, seg = t & 7;
        float m = m_sh[h], lsum = 0.f;
        for (int p = seg * 16; p < seg * 16 + 16; ++p) {
            float e = (p < nvalid) ? __expf(s_s[h * BS + p] - m) : 0.f;
            p_s[h * BS + p] = (__bf16)e;
            lsum += e;
        }
        red_s[h * 8 + seg] = lsum;
    }
    __syncthreads();
    if (t < NH) {
        float lsum = 0.f;
        for (int i = 0; i < 8; ++i) lsum += red_s[t * 8 + i];
        l_part[(b * BPS + blk) * NH + t] = lsum;
    }

    // prime V pipeline (half-chunk 0) — barrier below also covers p_s/l writes
    VStage vst;
    v_issue(kbase, latb, 0, t, haveCur, curRow, vst);
    v_commit(kv_s, t, vst);
    __syncthreads();

    // ---- ctx: C(16xLC) = P(16x128) x V(128xLC); 8 half-chunks (32 rows x 256 dims),
    //      double-buffered, 2 n-tiles per wave per iteration ----
    v8f cacc[4] = {{}, {}, {}, {}};
    for (int pc = 0; pc < 4; ++pc) {
        v16bf a;                                      // A fragment reused for both halves
        #pragma unroll
        for (int j = 0; j < 16; ++j)
            a[j] = p_s[row * BS + pc * 32 + kslot(lane, j)];
        #pragma unroll
        for (int half = 0; half < 2; ++half) {
            int it = pc * 2 + half;
            const __bf16* vbuf = kv_s + (it & 1) * 8192;
            VStage nx;
            if (it + 1 < 8) v_issue(kbase, latb, it + 1, t, haveCur, curRow, nx);
            #pragma unroll
            for (int i = 0; i < 2; ++i) {
                int dloc = (wave * 2 + i) * 16 + row; // dim within 256-wide half
                v16bf bm;
                #pragma unroll
                for (int j = 0; j < 16; ++j)
                    bm[j] = vbuf[kslot(lane, j) * 256 + dloc];
                cacc[half * 2 + i] = __builtin_amdgcn_wmma_f32_16x16x32_bf16(
                    false, a, false, bm, (short)0, cacc[half * 2 + i], false, false);
            }
            if (it + 1 < 8) v_commit(kv_s + ((it + 1) & 1) * 8192, t, nx);
            __syncthreads();
        }
    }
    #pragma unroll
    for (int half = 0; half < 2; ++half)
        #pragma unroll
        for (int i = 0; i < 2; ++i) {
            int nt = half * 16 + wave * 2 + i;        // global 16-wide n-tile
            #pragma unroll
            for (int r = 0; r < 8; ++r)
                ctxp[(size_t)(r + mo) * LC + nt * 16 + row] = cacc[half * 2 + i][r];
        }
}

// ---------------- Kernel 3: split-KV combine ----------------
__global__ __launch_bounds__(256) void reduce_kernel(
    const float* __restrict__ ctx_part, const float* __restrict__ m_part,
    const float* __restrict__ l_part, float* __restrict__ ctx_final) {
    int b = blockIdx.x, h = blockIdx.y, t = threadIdx.x;
    __shared__ float w_sh[BPS];
    __shared__ float Lsh;
    if (t == 0) {
        float M = NEG_INF;
        for (int s = 0; s < BPS; ++s) M = fmaxf(M, m_part[(b * BPS + s) * NH + h]);
        float Lt = 0.f;
        for (int s = 0; s < BPS; ++s) {
            float ms = m_part[(b * BPS + s) * NH + h];
            float w = (ms == NEG_INF) ? 0.f : __expf(ms - M);
            w_sh[s] = w;
            Lt += w * l_part[(b * BPS + s) * NH + h];
        }
        Lsh = Lt;
    }
    __syncthreads();
    float inv = 1.f / Lsh;
    for (int d = t; d < LC; d += 256) {
        float acc = 0.f;
        for (int s = 0; s < BPS; ++s)
            acc += w_sh[s] * ctx_part[(size_t)((b * BPS + s) * NH + h) * LC + d];
        ctx_final[(size_t)(b * NH + h) * LC + d] = acc * inv;
    }
}

// ---------------- Kernel 4: out = ctx . W_UV ----------------
__global__ __launch_bounds__(128) void out_kernel(
    const float* __restrict__ ctx_final, const float* __restrict__ W_UV,
    float* __restrict__ out) {
    int b = blockIdx.x, h = blockIdx.y, v = threadIdx.x;
    const float* c = ctx_final + (size_t)(b * NH + h) * LC;
    const float* W = W_UV + (size_t)h * LC * DV;
    float acc = 0.f;
    for (int l = 0; l < LC; ++l) acc += c[l] * W[(size_t)l * DV + v];
    out[(size_t)b * (NH * DV) + h * DV + v] = acc;
}

extern "C" void kernel_launch(void* const* d_in, const int* in_sizes, int n_in,
                              void* d_out, int out_size, void* d_ws, size_t ws_size,
                              hipStream_t stream) {
    const float* q      = (const float*)d_in[0];
    const float* k_c    = (const float*)d_in[1];
    const float* k_pe   = (const float*)d_in[2];
    const float* kvc    = (const float*)d_in[3];
    const float* W_UK_T = (const float*)d_in[4];
    const float* W_UV   = (const float*)d_in[5];
    const int* in_pos   = (const int*)d_in[6];
    // d_in[7] = slot_mapping (virtualized via current-token override)
    const int* btab     = (const int*)d_in[8];
    const int* slens    = (const int*)d_in[9];
    float* out          = (float*)d_out;

    char* ws = (char*)d_ws;
    size_t off = 0;
    auto alloc = [&](size_t bytes) {
        size_t p = off;
        off += (bytes + 255) & ~(size_t)255;
        return p;
    };
    __bf16* query_ws  = (__bf16*)(ws + alloc((size_t)NB * NH * DK * 2));
    __bf16* latent_ws = (__bf16*)(ws + alloc((size_t)NB * DK * 2));
    float*  ctx_part  = (float*)(ws + alloc((size_t)NB * BPS * NH * LC * 4));
    float*  m_part    = (float*)(ws + alloc((size_t)NB * BPS * NH * 4));
    float*  l_part    = (float*)(ws + alloc((size_t)NB * BPS * NH * 4));
    float*  ctx_final = (float*)(ws + alloc((size_t)NB * NH * LC * 4));

    prep_kernel<<<dim3(NB, NH), 256, 0, stream>>>(q, k_c, k_pe, W_UK_T, in_pos,
                                                  query_ws, latent_ws);
    attn_partial_kernel<<<dim3(NB, BPS), 256, 0, stream>>>(
        kvc, btab, slens, query_ws, latent_ws, ctx_part, m_part, l_part);
    reduce_kernel<<<dim3(NB, NH), 256, 0, stream>>>(ctx_part, m_part, l_part,
                                                    ctx_final);
    out_kernel<<<dim3(NB, NH), 128, 0, stream>>>(ctx_final, W_UV, out);
}
